// Sequence_43430709297205
// MI455X (gfx1250) — matrix operand order repaired and checked by reading, not compiled, not run on hardware
//
#include <hip/hip_runtime.h>
#include <hip/hip_bf16.h>
#include <cstdint>

#define HID    128
#define GATES  512
#define INPD   4
#define TSTEPS 2048
#define BATCH  2048
#define HS     136            // padded stride (elements) for h buffers only
#define HBUF   (16 * HS)      // one h buffer (elements)
#define WAVES  8
#define BLOCK  (WAVES * 32)
#define FRAG_E 512            // elements per 16x32 bf16 fragment (1 KB)

typedef __attribute__((ext_vector_type(16))) __bf16 v16bf;
typedef __attribute__((ext_vector_type(8)))  float  v8f;

union FragB {
    uint4 q[2];
    v16bf v;
    unsigned short s[16];
};

__device__ __forceinline__ unsigned short f2bf(float f) {
    unsigned int u = __float_as_uint(f);
    u += 0x7FFFu + ((u >> 16) & 1u);         // round-to-nearest-even
    return (unsigned short)(u >> 16);
}
__device__ __forceinline__ float sigm(float x)      { return 1.0f / (1.0f + __expf(-x)); }
__device__ __forceinline__ float tanh_fast(float x) { return 2.0f / (1.0f + __expf(-2.0f * x)) - 1.0f; }

__device__ __forceinline__ v8f splat8(float b) {
    v8f r = {b, b, b, b, b, b, b, b};
    return r;
}

__device__ __forceinline__ v16bf load_frag_pair(const unsigned short* p0,
                                                const unsigned short* p1) {
    FragB f;
    f.q[0] = *(const uint4*)p0;
    f.q[1] = *(const uint4*)p1;
    return f.v;
}

// A fragment (16x32 bf16) from h buffer [16 rows x HID, stride HS], chunk c.
__device__ __forceinline__ v16bf loadA(const unsigned short* hbase, int c) {
    const unsigned short* p = hbase + 32 * c;
    return load_frag_pair(p, p + 8);
}

// B fragment from fragment-linear storage: base already includes lane*16; 'off' in elements.
__device__ __forceinline__ v16bf loadBf(const unsigned short* base, int off) {
    const unsigned short* p = base + off;
    return load_frag_pair(p, p + 8);
}

#define WMMA_BF16(A, B, C) \
    __builtin_amdgcn_wmma_f32_16x16x32_bf16(false, (A), false, (B), (short)0, (C), false, false)

// Fragment-linear destination index for weight element (n, k) of a [512x128] matrix.
__device__ __forceinline__ int frag_dst(int n, int k) {
    int g   = n >> 7;
    int nn  = n & 127;
    int wv  = nn >> 4;
    int lmm = nn & 15;
    int c   = k >> 5;
    int kk  = k & 31;
    int hii = kk >> 4;
    int j   = kk & 15;
    return (((wv << 4) + (c << 2) + g) << 9) + (((hii << 4) + lmm) << 4) + j;
}

// ---------- prep: W_hh2 f32 -> bf16, fragment-linear ----------
__global__ void pack_whh2(const float* __restrict__ w, unsigned short* __restrict__ o) {
    int i = blockIdx.x * 256 + threadIdx.x;
    if (i >= GATES * HID) return;
    o[frag_dst(i >> 7, i & 127)] = f2bf(w[i]);
}

// ---------- persistent 2-layer LSTM over the full sequence ----------
__global__ void __launch_bounds__(BLOCK)
lstm_seq_kernel(const float* __restrict__ x,        // [T,B,4]
                const float* __restrict__ Wih1,     // [512,4]
                const float* __restrict__ Whh1,     // [512,128]
                const float* __restrict__ bih1,
                const float* __restrict__ bhh1,
                const float* __restrict__ Wih2,     // [512,128]
                const float* __restrict__ bih2,
                const float* __restrict__ bhh2,
                const unsigned short* __restrict__ whh2bf, // fragment-linear bf16
                const float* __restrict__ Wlin,     // [4,128]
                const float* __restrict__ blin,     // [4]
                float* __restrict__ out)            // [T,B,4]
{
    extern __shared__ __align__(16) unsigned char smem[];
    unsigned short* wh1   = (unsigned short*)smem;   // 65536 elems, fragment-linear (W_hh1)
    unsigned short* wi2   = wh1 + GATES * HID;       // 65536 elems, fragment-linear (W_ih2)
    unsigned short* h1bd  = wi2 + GATES * HID;       // 2 * HBUF (double-buffered h1)
    unsigned short* h2bd  = h1bd + 2 * HBUF;         // 2 * HBUF (double-buffered h2)
    unsigned short* wlinf = h2bd + 2 * HBUF;         // 4 frags * 512 elems (W_lin padded)

    const int tid  = threadIdx.x;
    const int lane = tid & 31;
    const int wave = tid >> 5;
    const int lm   = lane & 15;
    const int hi   = lane >> 4;
    const int b0   = blockIdx.x * 16;

    // --- stage weights into LDS (f32 -> bf16, fragment-linear) ---
    for (int i = tid; i < GATES * HID; i += BLOCK) {
        int dst = frag_dst(i >> 7, i & 127);
        wh1[dst] = f2bf(Whh1[i]);
        wi2[dst] = f2bf(Wih2[i]);
    }
    for (int i = tid; i < 4 * FRAG_E; i += BLOCK) {      // W_lin padded 4->16 cols
        int c = i >> 9, r = i & 511;
        int l = r >> 4, j = r & 15;
        int lmm = l & 15, hii = l >> 4;
        int k = 32 * c + hii * 16 + j;
        wlinf[i] = (lmm < INPD) ? f2bf(Wlin[lmm * HID + k]) : (unsigned short)0;
    }
    for (int i = tid; i < 2 * HBUF; i += BLOCK) { h1bd[i] = 0; h2bd[i] = 0; }

    // --- per-lane, time-invariant setup ---
    const int hsb = wave * 16;
    float bias1v[4], bias2v[4];
    FragB wx[4];                          // zero-padded K=32 B-fragments of W_ih1
    #pragma unroll
    for (int g = 0; g < 4; ++g) {
        int n = hsb + 128 * g + lm;
        bias1v[g] = bih1[n] + bhh1[n];
        bias2v[g] = bih2[n] + bhh2[n];
        wx[g].q[0] = make_uint4(0, 0, 0, 0);
        wx[g].q[1] = make_uint4(0, 0, 0, 0);
        if (hi == 0) {
            #pragma unroll
            for (int k = 0; k < INPD; ++k) wx[g].s[k] = f2bf(Wih1[n * INPD + k]);
        }
    }
    const float blin_l = blin[lm & (INPD - 1)];
    v8f c1 = splat8(0.0f);
    v8f c2 = splat8(0.0f);
    const v8f zeroacc = splat8(0.0f);     // folds to inline SRC2 = 0

    // base pointers; all fragment loads are immediate offsets from these
    const unsigned short* wh1b = wh1    + wave * (16 * FRAG_E) + lane * 16;
    const unsigned short* wi2b = wi2    + wave * (16 * FRAG_E) + lane * 16;
    const unsigned short* wh2b = whh2bf + wave * (16 * FRAG_E) + lane * 16;
    const unsigned short* wlb  = wlinf  + lane * 16;
    const unsigned short* h1ab = h1bd + lm * HS + hi * 16;  // A-frag lane base (h1)
    const unsigned short* h2ab = h2bd + lm * HS + hi * 16;  // A-frag lane base (h2)
    const int colw = hsb + lm;                              // h store column

    __syncthreads();

    for (int t = 0; t < TSTEPS; ++t) {
        const int po = (t & 1) * HBUF;    // read-buffer offset (t parity)
        const int wo = HBUF - po;         // write-buffer offset

        v8f acc1[4], acc2[4];

        // ---------------- layer 1: x_t part ----------------
        FragB ax;                         // x_t as zero-padded K=32 A fragment
        ax.q[0] = make_uint4(0, 0, 0, 0);
        ax.q[1] = make_uint4(0, 0, 0, 0);
        if (hi == 0) {
            const float* xp = x + ((size_t)t * BATCH + b0 + lm) * INPD;
            ax.s[0] = f2bf(xp[0]); ax.s[1] = f2bf(xp[1]);
            ax.s[2] = f2bf(xp[2]); ax.s[3] = f2bf(xp[3]);
            if (t + 1 < TSTEPS)
                __builtin_prefetch(xp + (size_t)BATCH * INPD, 0, 1);
        }
        #pragma unroll
        for (int g = 0; g < 4; ++g)
            acc1[g] = WMMA_BF16(ax.v, wx[g].v, zeroacc);

        // ---------------- layer 1: h1_{t-1} @ W_hh1 (LDS) ----------------
        #pragma unroll
        for (int c = 0; c < 4; ++c) {
            v16bf a = loadA(h1ab + po, c);
            v16bf b[4];
            #pragma unroll
            for (int g = 0; g < 4; ++g) b[g] = loadBf(wh1b, (c * 4 + g) * FRAG_E);
            #pragma unroll
            for (int g = 0; g < 4; ++g) acc1[g] = WMMA_BF16(a, b[g], acc1[g]);
        }

        // ------- layer 2 partial: h2_{t-1} @ W_hh2 (global, L2-resident) -------
        // Independent of h1_new -> computed before the h1 exchange; t-rotated
        // chunk order keeps global addresses loop-variant (no LICM hoisting).
        #pragma unroll
        for (int c0 = 0; c0 < 4; ++c0) {
            int c = (c0 + t) & 3;
            v16bf a = loadA(h2ab + po, c);
            const unsigned short* cb = wh2b + c * (4 * FRAG_E);
            v16bf b[4];
            #pragma unroll
            for (int g = 0; g < 4; ++g) b[g] = loadBf(cb, g * FRAG_E);
            #pragma unroll
            for (int g = 0; g < 4; ++g)
                acc2[g] = WMMA_BF16(a, b[g], (c0 == 0) ? zeroacc : acc2[g]);
        }

        // ---------------- elementwise cell update (layer 1) ----------------
        v8f h1v;
        #pragma unroll
        for (int e = 0; e < 8; ++e) {
            float iv = sigm(acc1[0][e] + bias1v[0]);
            float fv = sigm(acc1[1][e] + bias1v[1]);
            float gv = tanh_fast(acc1[2][e] + bias1v[2]);
            float ov = sigm(acc1[3][e] + bias1v[3]);
            float cn = fv * c1[e] + iv * gv;
            c1[e]  = cn;
            h1v[e] = ov * tanh_fast(cn);
        }
        // store h1_t into the write buffer (its last readers are 2 barriers back)
        #pragma unroll
        for (int e = 0; e < 8; ++e)
            h1bd[wo + (e + hi * 8) * HS + colw] = f2bf(h1v[e]);

        __syncthreads();                  // barrier A: h1_t visible

        // ---------------- layer 2 remainder: h1_t @ W_ih2 (LDS) ----------------
        #pragma unroll
        for (int c = 0; c < 4; ++c) {
            v16bf a = loadA(h1ab + wo, c);
            v16bf b[4];
            #pragma unroll
            for (int g = 0; g < 4; ++g) b[g] = loadBf(wi2b, (c * 4 + g) * FRAG_E);
            #pragma unroll
            for (int g = 0; g < 4; ++g) acc2[g] = WMMA_BF16(a, b[g], acc2[g]);
        }

        // ---------------- elementwise cell update (layer 2) ----------------
        v8f h2v;
        #pragma unroll
        for (int e = 0; e < 8; ++e) {
            float iv = sigm(acc2[0][e] + bias2v[0]);
            float fv = sigm(acc2[1][e] + bias2v[1]);
            float gv = tanh_fast(acc2[2][e] + bias2v[2]);
            float ov = sigm(acc2[3][e] + bias2v[3]);
            float cn = fv * c2[e] + iv * gv;
            c2[e]  = cn;
            h2v[e] = ov * tanh_fast(cn);
        }
        #pragma unroll
        for (int e = 0; e < 8; ++e)
            h2bd[wo + (e + hi * 8) * HS + colw] = f2bf(h2v[e]);

        __syncthreads();                  // barrier B: h2_t visible

        // -------- projection via WMMA: out = h2_t @ Wlin_pad + b_lin --------
        if (wave == 0) {                  // wave-uniform branch: EXEC all-1s for WMMA
            v8f po_acc;
            #pragma unroll
            for (int c = 0; c < 4; ++c) {
                v16bf a = loadA(h2ab + wo, c);
                v16bf b = loadBf(wlb, c * FRAG_E);
                po_acc = WMMA_BF16(a, b, (c == 0) ? zeroacc : po_acc);
            }
            if (lm < INPD) {
                float* op = out + ((size_t)t * BATCH + b0 + hi * 8) * INPD + lm;
                #pragma unroll
                for (int e = 0; e < 8; ++e)
                    op[(size_t)e * INPD] = po_acc[e] + blin_l;
            }
        }
    }
}

extern "C" void kernel_launch(void* const* d_in, const int* in_sizes, int n_in,
                              void* d_out, int out_size, void* d_ws, size_t ws_size,
                              hipStream_t stream) {
    (void)in_sizes; (void)n_in; (void)out_size; (void)ws_size;

    const float* input = (const float*)d_in[0];
    // d_in[1] = future (unused, == 0)
    const float* Wih1  = (const float*)d_in[2];
    const float* Whh1  = (const float*)d_in[3];
    const float* bih1  = (const float*)d_in[4];
    const float* bhh1  = (const float*)d_in[5];
    const float* Wih2  = (const float*)d_in[6];
    const float* Whh2  = (const float*)d_in[7];
    const float* bih2  = (const float*)d_in[8];
    const float* bhh2  = (const float*)d_in[9];
    const float* Wlin  = (const float*)d_in[10];
    const float* blin  = (const float*)d_in[11];
    // d_in[12..15] = h0,c0,h02,c02 (all zeros; kernel re-zeros its state itself)

    unsigned short* whh2bf = (unsigned short*)d_ws;

    pack_whh2<<<(GATES * HID + 255) / 256, 256, 0, stream>>>(Whh2, whh2bf);

    size_t shmem = (size_t)(2 * GATES * HID + 4 * HBUF + 4 * FRAG_E)
                 * sizeof(unsigned short);     // = 283,648 bytes (< 320 KB WGP LDS)

    lstm_seq_kernel<<<BATCH / 16, BLOCK, shmem, stream>>>(
        input, Wih1, Whh1, bih1, bhh1, Wih2, bih2, bhh2,
        whh2bf, Wlin, blin, (float*)d_out);
}
